// SuperKANLinear_54382875902123
// MI455X (gfx1250) — compile-verified, matrix operand order checked by reading, not run on hardware
//
#include <hip/hip_runtime.h>
#include <hip/hip_bf16.h>
#include <stdint.h>

// ---------------------------------------------------------------------------
// SuperKANLinear fused kernel for gfx1250 (MI455X), wave32 + WMMA bf16.
//   out[b,o] = gate[o]*spline[b,o] + (1-gate[o])*(base[b,o] + base_b[o])
// Gate is folded into the B-matrix at LDS fill time -> single f32 accumulator.
// ---------------------------------------------------------------------------

typedef __attribute__((ext_vector_type(16))) __bf16 v16bf;
typedef __attribute__((ext_vector_type(8)))  float  v8f;

#define IN_F   2048
#define OUT_F  2048
#define NG     8
#define BATCH  4096

#define MT 64               // batch rows per block
#define NT 32               // out features per block
#define KI 32               // input features per K-iteration
#define KS (KI*NG)          // spline K per iteration = 256
#define NITER (IN_F/KI)     // 64

// LDS layout (byte offsets). Rows padded to kill ds bank conflicts.
#define ASPL_STRIDE 528     // 256 bf16 (512B) + 16B pad  -> lane l starts at bank 4l
#define ABASE_STRIDE 80     // 32 bf16 (64B) + 16B pad
#define OFF_ASPL  0
#define OFF_BSPL  (OFF_ASPL  + MT*ASPL_STRIDE)    // 33792
#define OFF_ABASE (OFF_BSPL  + NT*ASPL_STRIDE)    // 50688
#define OFF_BBASE (OFF_ABASE + MT*ABASE_STRIDE)   // 55808
#define LDS_BYTES (OFF_BBASE + NT*ABASE_STRIDE)   // 58368

union FragB { v16bf h; uint4 q[2]; uint2 d[4]; };

__device__ __forceinline__ unsigned short f2bf(float f) {
  unsigned u = __float_as_uint(f);
  u += 0x7FFFu + ((u >> 16) & 1u);           // round-to-nearest-even
  return (unsigned short)(u >> 16);
}
__device__ __forceinline__ float sigmoidf_(float v) {
  return 1.0f / (1.0f + __expf(-v));
}

__global__ __launch_bounds__(256)
void superkan_wmma_kernel(const float* __restrict__ x,
                          const float* __restrict__ base_w,
                          const float* __restrict__ base_b,
                          const float* __restrict__ spline_w,
                          const float* __restrict__ grid,
                          const float* __restrict__ adaptive_gate,
                          const float* __restrict__ feat_imp,
                          float* __restrict__ out)
{
  __shared__ uint4 smem4[LDS_BYTES / 16];
  char* lds = (char*)smem4;

  const int tid = (int)threadIdx.x;
  const int bm  = (int)blockIdx.x;   // batch block   (64 blocks)
  const int bn  = (int)blockIdx.y;   // out-feat block(64 blocks)

  // grid points (8 floats) into registers once
  float gr[8];
  {
    const float4 g0 = ((const float4*)grid)[0];
    const float4 g1 = ((const float4*)grid)[1];
    gr[0]=g0.x; gr[1]=g0.y; gr[2]=g0.z; gr[3]=g0.w;
    gr[4]=g1.x; gr[5]=g1.y; gr[6]=g1.z; gr[7]=g1.w;
  }

  // wave mapping: 8 waves, each owns one 16x16 output tile
  const int wave = tid >> 5;
  const int lane = tid & 31;
  const int wm   = wave & 3;     // M subtile 0..3
  const int wn   = wave >> 2;    // N subtile 0..1
  const int lrow = lane & 15;
  const int half = lane >> 4;

  v8f acc = {};                  // f32 accumulator (gate pre-folded into B)

  // fill-phase mapping
  const int a_row   = tid >> 2;          // 0..63  batch row in tile
  const int a_iseg  = (tid & 3) * 8;     // 8 input features each
  const int b_row   = tid >> 3;          // 0..31  out row in tile
  const int bb_iseg = (tid & 7) * 4;     // base_w: 4 floats each
  const int bs_kseg = (tid & 7) * 32;    // spline: 32 floats each

  const int   o_glob = bn * NT + b_row;
  const float gate_b = sigmoidf_(adaptive_gate[o_glob]);
  const float sc_base = 1.0f - gate_b;

  for (int ki = 0; ki < NITER; ++ki) {
    // ---------------- fill LDS ----------------
    // A tiles: xw (base) + normalized RBF basis (spline), converted to bf16
    {
      const float* xp = x + (size_t)(bm*MT + a_row)*IN_F + ki*KI + a_iseg;
      const float* fp = feat_imp + ki*KI + a_iseg;
      float4 x0 = ((const float4*)xp)[0];
      float4 x1 = ((const float4*)xp)[1];
      float4 f0 = ((const float4*)fp)[0];
      float4 f1 = ((const float4*)fp)[1];
      float xv[8] = {x0.x,x0.y,x0.z,x0.w,x1.x,x1.y,x1.z,x1.w};
      float fv[8] = {f0.x,f0.y,f0.z,f0.w,f1.x,f1.y,f1.z,f1.w};
      unsigned short ab[8];
      #pragma unroll
      for (int e = 0; e < 8; ++e) {
        float xw = xv[e] * sigmoidf_(fv[e]);
        ab[e] = f2bf(xw);
        float xn = tanhf(xw);
        float bas[8]; float sum = 0.f;
        #pragma unroll
        for (int g = 0; g < 8; ++g) {
          float d = xn - gr[g];
          float bv = __expf(-2.0f * d * d);
          bas[g] = bv; sum += bv;
        }
        float inv = 1.0f / fmaxf(sum, 1e-12f);
        unsigned short hb[8];
        #pragma unroll
        for (int g = 0; g < 8; ++g) hb[g] = f2bf(bas[g] * inv);
        uint4 pk;
        pk.x = (unsigned)hb[0] | ((unsigned)hb[1] << 16);
        pk.y = (unsigned)hb[2] | ((unsigned)hb[3] << 16);
        pk.z = (unsigned)hb[4] | ((unsigned)hb[5] << 16);
        pk.w = (unsigned)hb[6] | ((unsigned)hb[7] << 16);
        *(uint4*)(lds + OFF_ASPL + a_row*ASPL_STRIDE + (a_iseg + e)*16) = pk;
      }
      uint2 p0, p1;
      p0.x = (unsigned)ab[0] | ((unsigned)ab[1] << 16);
      p0.y = (unsigned)ab[2] | ((unsigned)ab[3] << 16);
      p1.x = (unsigned)ab[4] | ((unsigned)ab[5] << 16);
      p1.y = (unsigned)ab[6] | ((unsigned)ab[7] << 16);
      char* abp = lds + OFF_ABASE + a_row*ABASE_STRIDE + a_iseg*2;
      *(uint2*)(abp)     = p0;
      *(uint2*)(abp + 8) = p1;
    }
    // B base tile: (1-gate[o]) * base_w[o, ki*32 .. +32)
    {
      const float* wp = base_w + (size_t)o_glob*IN_F + ki*KI + bb_iseg;
      float4 w = ((const float4*)wp)[0];
      uint2 p;
      p.x = (unsigned)f2bf(w.x*sc_base) | ((unsigned)f2bf(w.y*sc_base) << 16);
      p.y = (unsigned)f2bf(w.z*sc_base) | ((unsigned)f2bf(w.w*sc_base) << 16);
      *(uint2*)(lds + OFF_BBASE + b_row*ABASE_STRIDE + bb_iseg*2) = p;
    }
    // B spline tile: gate[o] * spline_w[o, ki*256 .. +256) (i,g contiguous = K)
    {
      const float* sp = spline_w + (size_t)o_glob*(IN_F*NG) + ki*KS + bs_kseg;
      #pragma unroll
      for (int q = 0; q < 4; ++q) {
        float4 wa = ((const float4*)sp)[q*2 + 0];
        float4 wb = ((const float4*)sp)[q*2 + 1];
        uint4 pk;
        pk.x = (unsigned)f2bf(wa.x*gate_b) | ((unsigned)f2bf(wa.y*gate_b) << 16);
        pk.y = (unsigned)f2bf(wa.z*gate_b) | ((unsigned)f2bf(wa.w*gate_b) << 16);
        pk.z = (unsigned)f2bf(wb.x*gate_b) | ((unsigned)f2bf(wb.y*gate_b) << 16);
        pk.w = (unsigned)f2bf(wb.z*gate_b) | ((unsigned)f2bf(wb.w*gate_b) << 16);
        *(uint4*)(lds + OFF_BSPL + b_row*ASPL_STRIDE + bs_kseg*2 + q*16) = pk;
      }
      if (ki + 1 < NITER) __builtin_prefetch(sp + KS, 0, 1);  // global_prefetch_b8
    }
    __syncthreads();

    // ---------------- compute ----------------
    // Fragment layout (16-bit A 16x32, ISA 7.12.2): lanes 0-15 -> K[0..7],K[16..23],
    // lanes 16-31 -> K[8..15],K[24..31]; B tile stored [N][K] uses identical pattern.
    {
      // base path: one K=32 WMMA
      FragB a, b;
      {
        const char* ap = lds + OFF_ABASE + (wm*16 + lrow)*ABASE_STRIDE + half*16;
        a.d[0] = *(const uint2*)(ap);
        a.d[1] = *(const uint2*)(ap + 8);
        a.d[2] = *(const uint2*)(ap + 32);
        a.d[3] = *(const uint2*)(ap + 40);
        const char* bp = lds + OFF_BBASE + (wn*16 + lrow)*ABASE_STRIDE + half*16;
        b.d[0] = *(const uint2*)(bp);
        b.d[1] = *(const uint2*)(bp + 8);
        b.d[2] = *(const uint2*)(bp + 32);
        b.d[3] = *(const uint2*)(bp + 40);
      }
      acc = __builtin_amdgcn_wmma_f32_16x16x32_bf16(
                false, a.h, false, b.h, (short)0, acc, false, false);

      // spline path: 8 K=32 WMMAs over the 256-wide chunk
      const char* asp = lds + OFF_ASPL + (wm*16 + lrow)*ASPL_STRIDE + half*16;
      const char* bsp = lds + OFF_BSPL + (wn*16 + lrow)*ASPL_STRIDE + half*16;
      #pragma unroll
      for (int ks = 0; ks < 8; ++ks) {
        FragB as, bs;
        as.q[0] = *(const uint4*)(asp + ks*64);
        as.q[1] = *(const uint4*)(asp + ks*64 + 32);
        bs.q[0] = *(const uint4*)(bsp + ks*64);
        bs.q[1] = *(const uint4*)(bsp + ks*64 + 32);
        acc = __builtin_amdgcn_wmma_f32_16x16x32_bf16(
                  false, as.h, false, bs.h, (short)0, acc, false, false);
      }
    }
    __syncthreads();
  }

  // ---------------- epilogue ----------------
  // D layout: VGPR r, lanes 0-15 -> (M=r, N=lane), lanes 16-31 -> (M=r+8, N=lane-16)
  {
    const int   n    = bn*NT + wn*16 + lrow;
    const float gate = sigmoidf_(adaptive_gate[n]);
    const float bias = (1.0f - gate) * base_b[n];
    const int   m0   = bm*MT + wm*16 + half*8;
    #pragma unroll
    for (int r = 0; r < 8; ++r) {
      out[(size_t)(m0 + r)*OUT_F + n] = acc[r] + bias;
    }
  }
}

extern "C" void kernel_launch(void* const* d_in, const int* in_sizes, int n_in,
                              void* d_out, int out_size, void* d_ws, size_t ws_size,
                              hipStream_t stream) {
  (void)in_sizes; (void)n_in; (void)out_size; (void)d_ws; (void)ws_size;
  const float* x     = (const float*)d_in[0];
  const float* bw    = (const float*)d_in[1];
  const float* bb    = (const float*)d_in[2];
  const float* sw    = (const float*)d_in[3];
  const float* grid  = (const float*)d_in[4];
  const float* gate  = (const float*)d_in[5];
  const float* fi    = (const float*)d_in[6];
  float* out = (float*)d_out;

  dim3 gridDim(BATCH / MT, OUT_F / NT);   // (64, 64); x-major so consecutive
                                          // blocks share the same weight slice (L2)
  superkan_wmma_kernel<<<gridDim, 256, 0, stream>>>(x, bw, bb, sw, grid, gate, fi, out);
}
